// MSDeformAttn_63471026700779
// MI455X (gfx1250) — matrix operand order repaired and checked by reading, not compiled, not run on hardware
//
#include <hip/hip_runtime.h>
#include <hip/hip_bf16.h>

typedef __attribute__((ext_vector_type(16))) __bf16 v16bf;
typedef __attribute__((ext_vector_type(8)))  float  v8f;

#define D_MODEL  256
#define N_HEADS  8
#define DH       32
#define LEN_V    13294
#define BS       4
#define MROWS    (BS * LEN_V)   // 53176

__device__ __forceinline__ __bf16 f2bf(float f)  { return (__bf16)f; }
__device__ __forceinline__ float  bf2f(__bf16 b) { return (float)b; }

union FragHalfPair { float4 q[2]; v16bf v; };

// =====================================================================
// Weight pre-pack: W(KxN) fp32 -> split-bf16 WMMA B-fragments.
// Fragment (kstep, ntile): lane<16 holds col N=n0+lane, K=k0+e;
// lane>=16 holds col n0+lane-16, K=k0+16+e   (e = 0..15).
// Packed: frag_idx = (kstep*tilesN + ntile)*32 + lane; 64B = hi(32B)|lo(32B).
// =====================================================================
template <int K, int N>
__global__ __launch_bounds__(256)
void pack_w_kernel(const float* __restrict__ Wm, __bf16* __restrict__ packed) {
    constexpr int tilesN = N / 16;
    constexpr int ksteps = K / 32;
    int tid = blockIdx.x * blockDim.x + threadIdx.x;
    if (tid >= ksteps * tilesN * 32) return;
    int lane  = tid & 31;
    int frag  = tid >> 5;
    int ntile = frag % tilesN;
    int kstep = frag / tilesN;
    int l16 = lane & 15, lh = lane >> 4;
    const float* p = Wm + (kstep * 32 + lh * 16) * N + ntile * 16 + l16;
    v16bf hi, lo;
#pragma unroll
    for (int e = 0; e < 16; ++e) {
        float f = p[e * N];
        __bf16 h = f2bf(f);
        hi[e] = h;
        lo[e] = f2bf(f - bf2f(h));
    }
    v16bf* dst = (v16bf*)packed;
    dst[(size_t)tid * 2 + 0] = hi;
    dst[(size_t)tid * 2 + 1] = lo;
}

// ---------- A fragment: fp32 -> split bf16 on the fly ----------
// lane<16: M=row0+lane, K elements 0..15; lane>=16: K elements 16..31.
__device__ __forceinline__ void load_a_frag(const float* __restrict__ A, int M, int K,
                                            int row0, int k0, v16bf& hi, v16bf& lo) {
    int lane = threadIdx.x & 31;
    int l16 = lane & 15, lh = lane >> 4;
    int row = row0 + l16; if (row >= M) row = M - 1;   // clamp keeps EXEC full
    const float* p = A + row * K + k0 + lh * 16;
    __builtin_prefetch(p + 32, 0, 0);                  // next k-step (global_prefetch_b8)
#pragma unroll
    for (int e = 0; e < 8; ++e) {
        float f0 = p[e];
        float f1 = p[8 + e];
        __bf16 h0 = f2bf(f0), h1 = f2bf(f1);
        hi[e] = h0;      hi[8 + e] = h1;
        lo[e]     = f2bf(f0 - bf2f(h0));
        lo[8 + e] = f2bf(f1 - bf2f(h1));
    }
}

// =====================================================================
// C = A(MxK) @ W(KxN) + bias.
// Block = 8 waves = 8 consecutive M-tiles, all on ONE 64-col N-group.
// The N-group's packed B fragments (all K) are staged once into LDS
// (64 KB, unit-major so consecutive lanes hit consecutive banks), then
// every wave consumes them via ds_load_b128.
// Split-bf16: hi*hi + lo*hi + hi*lo  (~fp32 accuracy).
// =====================================================================
template <int K, int N>
__global__ __launch_bounds__(256)
void gemm_bias_kernel(const float* __restrict__ A, const __bf16* __restrict__ packedB,
                      const float* __restrict__ bias, float* __restrict__ C, int M) {
    constexpr int tilesN = N / 16;
    constexpr int NG     = tilesN / 4;          // 64-col groups
    constexpr int ksteps = K / 32;
    __shared__ __bf16 sB[ksteps * 4 * 32 * 32]; // ksteps*4 frags * 32 lanes * 64B = 64KB

    int tilesM  = (M + 15) >> 4;
    int mblocks = (tilesM + 7) >> 3;
    int ng = blockIdx.x % NG;
    int mb = blockIdx.x / NG;
    if (mb >= mblocks) return;                  // block-uniform

    // ---- stage this N-group's B fragments into LDS (unit-major layout) ----
    // LDS float4 index: (f*4 + u)*32 + lane,  f = ks*4 + t, u = 0..3 (hi:0,1 lo:2,3)
    {
        const float4* src = (const float4*)packedB;
        float4* dst = (float4*)sB;
        for (int fl = threadIdx.x; fl < ksteps * 4 * 32; fl += 256) {
            int lane = fl & 31;
            int f    = fl >> 5;                 // ks*4 + t
            int ks   = f >> 2, t = f & 3;
            int g    = (ks * tilesN + ng * 4 + t) * 32 + lane;  // global frag-lane
#pragma unroll
            for (int u = 0; u < 4; ++u)
                dst[(f * 4 + u) * 32 + lane] = src[g * 4 + u];
        }
    }
    __syncthreads();

    int waveInBlk = threadIdx.x >> 5;
    int tm = mb * 8 + waveInBlk;
    bool active = tm < tilesM;
    if (!active) tm = tilesM - 1;               // clamp: keep wave in lockstep
    int row0 = tm << 4;
    int lane = threadIdx.x & 31;
    const float4* ls4 = (const float4*)sB;

    v8f acc0 = {}, acc1 = {}, acc2 = {}, acc3 = {};
#pragma unroll 2
    for (int ks = 0; ks < ksteps; ++ks) {
        v16bf ah, al;
        load_a_frag(A, M, K, row0, ks * 32, ah, al);
#pragma unroll
        for (int t = 0; t < 4; ++t) {
            int fi = ((ks * 4 + t) * 4) * 32 + lane;
            FragHalfPair H, L;
            H.q[0] = ls4[fi];          H.q[1] = ls4[fi + 32];
            L.q[0] = ls4[fi + 64];     L.q[1] = ls4[fi + 96];
            v16bf bh = H.v, bl = L.v;
            v8f a = (t == 0) ? acc0 : (t == 1) ? acc1 : (t == 2) ? acc2 : acc3;
            a = __builtin_amdgcn_wmma_f32_16x16x32_bf16(false, ah, false, bh, (short)0, a, false, false);
            a = __builtin_amdgcn_wmma_f32_16x16x32_bf16(false, al, false, bh, (short)0, a, false, false);
            a = __builtin_amdgcn_wmma_f32_16x16x32_bf16(false, ah, false, bl, (short)0, a, false, false);
            if (t == 0) acc0 = a; else if (t == 1) acc1 = a; else if (t == 2) acc2 = a; else acc3 = a;
        }
    }

    int l16 = lane & 15, lh = lane >> 4;
#pragma unroll
    for (int t = 0; t < 4; ++t) {
        int col = (ng * 4 + t) * 16 + l16;
        float b = bias[col];
        const v8f a = (t == 0) ? acc0 : (t == 1) ? acc1 : (t == 2) ? acc2 : acc3;
#pragma unroll
        for (int r = 0; r < 8; ++r) {
            int row = row0 + r + lh * 8;
            if (active && row < M) C[row * N + col] = a[r] + b;  // 32-bit index math
        }
    }
}

// =====================================================================
// Sampling core: one wave per (b,q,head); lane = channel (coalesced 128B
// gathers from v, which lives in L2). Uniform scalars via readfirstlane.
// =====================================================================
__global__ __launch_bounds__(256)
void msda_sample_kernel(const float* __restrict__ v, const float* __restrict__ off,
                        const float* __restrict__ aw, const float* __restrict__ bbox,
                        float* __restrict__ out) {
    const int lvlH[4] = {100, 50, 25, 13};
    const int lvlW[4] = {100, 50, 25, 13};
    const int lvlS[4] = {0, 10000, 12500, 13125};

    int task = blockIdx.x * (blockDim.x >> 5) + (threadIdx.x >> 5);
    const int total = BS * LEN_V * N_HEADS;
    if (task >= total) return;
    task = __builtin_amdgcn_readfirstlane(task);   // wave-uniform -> scalar loads

    int h  = task & 7;
    int bq = task >> 3;            // = b*LEN_V + q
    int b  = bq / LEN_V;
    int lane = threadIdx.x & 31;   // channel within head

    const float* awp = aw + (size_t)bq * (N_HEADS * 16) + h * 16;
    float mx = -1e30f;
#pragma unroll
    for (int i = 0; i < 16; ++i) mx = fmaxf(mx, awp[i]);
    float e[16], ssum = 0.f;
#pragma unroll
    for (int i = 0; i < 16; ++i) { e[i] = __expf(awp[i] - mx); ssum += e[i]; }
    float inv = 1.f / ssum;

    const float* offp  = off  + (size_t)bq * (N_HEADS * 32) + h * 32; // [lvl*8 + p*2 + xy]
    const float* bbp   = bbox + (size_t)bq * 8;                        // [lvl*2 + xy]
    const float* vbase = v + (size_t)b * LEN_V * D_MODEL + h * DH + lane;

    float acc = 0.f;
#pragma unroll
    for (int lvl = 0; lvl < 4; ++lvl) {
        const int Wi = lvlW[lvl], Hi = lvlH[lvl];
        const float Wf = (float)Wi, Hf = (float)Hi;
        const float* vl = vbase + (size_t)lvlS[lvl] * D_MODEL;
        float bx = bbp[lvl * 2 + 0], by = bbp[lvl * 2 + 1];
#pragma unroll
        for (int p = 0; p < 4; ++p) {
            float ox = offp[lvl * 8 + p * 2 + 0];
            float oy = offp[lvl * 8 + p * 2 + 1];
            float x = fmaf(bx, Wf, ox) - 0.5f;    // pix = bbox*dim + off - 0.5
            float y = fmaf(by, Hf, oy) - 0.5f;
            float x0f = floorf(x), y0f = floorf(y);
            float lx = x - x0f, ly = y - y0f;
            int x0 = (int)x0f, y0 = (int)y0f;
            float ew = e[lvl * 4 + p];
#pragma unroll
            for (int dy = 0; dy < 2; ++dy) {
#pragma unroll
                for (int dx = 0; dx < 2; ++dx) {
                    int xi = x0 + dx, yi = y0 + dy;
                    float wgt = (dx ? lx : 1.f - lx) * (dy ? ly : 1.f - ly);
                    bool valid = (xi >= 0) && (xi < Wi) && (yi >= 0) && (yi < Hi);
                    int xc = min(max(xi, 0), Wi - 1);
                    int yc = min(max(yi, 0), Hi - 1);
                    float g = vl[(yc * Wi + xc) * D_MODEL];  // coalesced 128B/wave
                    acc = fmaf(ew * (valid ? wgt : 0.f), g, acc);
                }
            }
        }
    }
    out[(size_t)bq * D_MODEL + h * DH + lane] = acc * inv;
}

extern "C" void kernel_launch(void* const* d_in, const int* in_sizes, int n_in,
                              void* d_out, int out_size, void* d_ws, size_t ws_size,
                              hipStream_t stream) {
    const float* query  = (const float*)d_in[0];
    const float* bbox   = (const float*)d_in[1];
    const float* value  = (const float*)d_in[2];
    const float* W_val  = (const float*)d_in[3];
    const float* b_val  = (const float*)d_in[4];
    const float* W_off  = (const float*)d_in[5];
    const float* b_off  = (const float*)d_in[6];
    const float* W_attn = (const float*)d_in[7];
    const float* b_attn = (const float*)d_in[8];
    const float* W_out  = (const float*)d_in[9];
    const float* b_out  = (const float*)d_in[10];
    float* out = (float*)d_out;

    // fp32 workspace tensors
    float* v_ws    = (float*)d_ws;                       // MROWS*256
    float* off_ws  = v_ws   + (size_t)MROWS * 256;       // MROWS*256
    float* aw_ws   = off_ws + (size_t)MROWS * 256;       // MROWS*128
    float* msda_ws = aw_ws  + (size_t)MROWS * 128;       // MROWS*256
    // packed split-bf16 weights (hi|lo per fragment)
    __bf16* pk_val  = (__bf16*)(msda_ws + (size_t)MROWS * 256);
    __bf16* pk_off  = pk_val  + 8 * 16 * 32 * 32;        // K/32 * N/16 * 32 lanes * 32 bf16
    __bf16* pk_attn = pk_off  + 8 * 16 * 32 * 32;
    __bf16* pk_out  = pk_attn + 8 *  8 * 32 * 32;

    const int M = MROWS;
    const int tilesM  = (M + 15) / 16;     // 3324
    const int mblocks = (tilesM + 7) / 8;  // 416
    dim3 blk(256);

    // --- pack weights into WMMA fragment layout (cheap, once per launch) ---
    pack_w_kernel<256, 256><<<dim3((8 * 16 * 32 + 255) / 256), blk, 0, stream>>>(W_val,  pk_val);
    pack_w_kernel<256, 256><<<dim3((8 * 16 * 32 + 255) / 256), blk, 0, stream>>>(W_off,  pk_off);
    pack_w_kernel<256, 128><<<dim3((8 *  8 * 32 + 255) / 256), blk, 0, stream>>>(W_attn, pk_attn);
    pack_w_kernel<256, 256><<<dim3((8 * 16 * 32 + 255) / 256), blk, 0, stream>>>(W_out,  pk_out);

    // --- projections: grid = mblocks * (N/64) ---
    gemm_bias_kernel<256, 256><<<dim3(mblocks * 4), blk, 0, stream>>>(value,   pk_val,  b_val,  v_ws,   M);
    gemm_bias_kernel<256, 256><<<dim3(mblocks * 4), blk, 0, stream>>>(query,   pk_off,  b_off,  off_ws, M);
    gemm_bias_kernel<256, 128><<<dim3(mblocks * 2), blk, 0, stream>>>(query,   pk_attn, b_attn, aw_ws,  M);

    // --- softmax + bilinear sampling + weighted accumulation ---
    const long totalTasks = (long)BS * LEN_V * N_HEADS;
    msda_sample_kernel<<<dim3((unsigned)((totalTasks + 7) / 8)), blk, 0, stream>>>(v_ws, off_ws, aw_ws, bbox, msda_ws);

    // --- output projection ---
    gemm_bias_kernel<256, 256><<<dim3(mblocks * 4), blk, 0, stream>>>(msda_ws, pk_out, b_out, out, M);
}
